// CausalSelfAttention_13168369730161
// MI455X (gfx1250) — compile-verified, matrix operand order checked by reading, not run on hardware
//
#include <hip/hip_runtime.h>

// ---------------------------------------------------------------------------
// CDNA5 (gfx1250) causal self-attention, bf16 WMMA pipeline with TDM:
//   1) convert x -> bf16, transpose+convert W_qkv, W_proj -> bf16 [N][K]
//   2) QKV GEMM: A/B k-tiles DMA'd into LDS via tensor_load_to_lds
//      (TENSORcnt double buffer), WMMA consumes from LDS via ds_load_b128;
//      scatter to Q [B][H][T][64], K [B][H][T][64], V^T [B][H][64][T] (bf16)
//   3) flash attention: per wave one 16-row q tile, online softmax,
//      scores & P@V via v_wmma_f32_16x16x32_bf16, P transposed via LDS
//   4) output GEMM: same TDM-fed WMMA mainloop, fp32 out + bias
// ---------------------------------------------------------------------------

typedef __attribute__((ext_vector_type(16))) __bf16        v16bf;
typedef __attribute__((ext_vector_type(8)))  float         v8f;
typedef __attribute__((ext_vector_type(4)))  unsigned int  v4u;
typedef __attribute__((ext_vector_type(4)))  float         v4f;
typedef __attribute__((ext_vector_type(4)))  unsigned int  u32x4;
typedef __attribute__((ext_vector_type(8)))  int           i32x8;
typedef __attribute__((ext_vector_type(4)))  int           i32x4;

union FragBF {
    v16bf v;
    v4u   q[2];
    unsigned short u[16];
};

// Escape the pointee (input-only asm operand + memory clobber) WITHOUT
// rewriting the pointer: alias analysis must then assume asm ::: "memory"
// fences (and the invisible TDM writes they stand in for) modify it, while
// the original pointer keeps its LDS provenance so loads stay ds_load_*.
template <typename T>
__device__ __forceinline__ void escape(T* p) {
    asm volatile("" :: "v"(p) : "memory");
}

__device__ __forceinline__ unsigned short f2bf(float f) {
    union { float f; unsigned int u; } cv; cv.f = f;
    unsigned int u = cv.u + 0x7FFFu + ((cv.u >> 16) & 1u);  // RNE
    return (unsigned short)(u >> 16);
}

// A/B fragment (16x32 bf16, row-major source, lane = row/col, two b128 loads)
__device__ __forceinline__ FragBF load_frag_row(const unsigned short* __restrict__ base,
                                                int row, int ld, int k0, int hi) {
    FragBF a;
    const unsigned short* p = base + row * ld + k0 + hi * 8;
    a.q[0] = *(const v4u*)(p);
    a.q[1] = *(const v4u*)(p + 16);
    return a;
}

// ---------------- Tensor Data Mover: async 2D tile -> LDS ------------------
// Builds a D# per CDNA5 ISA ch.8 and issues TENSOR_LOAD_TO_LDS (TENSORcnt).
// 2-byte elements; tile rows packed contiguously into LDS [tile_d1][tile_d0].
__device__ __forceinline__ void tdm_load_2d(unsigned lds_byte_addr,
                                            const unsigned short* gptr,
                                            unsigned tensor_d0, unsigned tensor_d1,
                                            unsigned tile_d0, unsigned tile_d1,
                                            unsigned stride0_elems) {
    unsigned long long ga = (unsigned long long)(uintptr_t)gptr;
    u32x4 g0;
    g0[0] = 1u;                                         // count=1 valid descriptor
    g0[1] = lds_byte_addr;                              // LDS dest (bytes)
    g0[2] = (unsigned)(ga & 0xFFFFFFFFull);             // global_addr[31:0]
    g0[3] = (unsigned)((ga >> 32) & 0x01FFFFFFull)      // global_addr[56:32]
          | 0x80000000u;                                // type=2 ("image")
    i32x8 g1;
    g1[0] = 0x00010000;                                 // data_size=1 -> 2 bytes
    g1[1] = (int)((tensor_d0 & 0xFFFFu) << 16);         // tensor_dim0[15:0]
    g1[2] = (int)(((tensor_d0 >> 16) & 0xFFFFu)         // tensor_dim0[31:16]
          | ((tensor_d1 & 0xFFFFu) << 16));             // tensor_dim1[15:0]
    g1[3] = (int)(((tensor_d1 >> 16) & 0xFFFFu)         // tensor_dim1[31:16]
          | (tile_d0 << 16));                           // tile_dim0
    g1[4] = (int)(tile_d1 & 0xFFFFu);                   // tile_dim1 (tile_dim2=0)
    g1[5] = (int)stride0_elems;                         // tensor_dim0_stride[31:0]
    g1[6] = 0;                                          // stride0 hi / dim1_stride lo
    g1[7] = 0;
    i32x4 z4 = {0, 0, 0, 0};                            // groups 2/3: 2-D tensor
    i32x8 z8 = {0, 0, 0, 0, 0, 0, 0, 0};
    // amdgpu-toolchain (clang-23) 6-arg form:
    //   (u32x4 g0, i32x8 g1, i32x4 g2, i32x4 g3, i32x8, i32 cpol)
    __builtin_amdgcn_tensor_load_to_lds(g0, g1, z4, z4, z8, 0);
}

// TDM-fed, TENSORcnt-double-buffered WMMA mainloop.
// Wave tile: 16(M) x 64(N); per buffer: B tile [64][32] (4KB) + A tile [16][32]
// (1KB) packed at lds_base + buf*5120 (B) / +4096 (A).
__device__ __forceinline__ void gemm_mainloop_tdm(const unsigned short* __restrict__ A,
                                                  const unsigned short* __restrict__ Bt,
                                                  const unsigned short* lds_base,
                                                  unsigned lds_off, int m0, int n0,
                                                  int K, int Mt, int Nt,
                                                  int ln, int hi, v8f acc[4]) {
    escape(lds_base);   // TDM writes this buffer behind the compiler's back

    tdm_load_2d(lds_off,        Bt + (unsigned)n0 * K, K, Nt, 32, 64, K);
    tdm_load_2d(lds_off + 4096, A  + (unsigned)m0 * K, K, Mt, 32, 16, K);

    int buf = 0;
    for (int k0 = 0; k0 < K; k0 += 32) {
        if (k0 + 32 < K) {
            // previous reads of buf^1 must finish before the DMA overwrites it
            asm volatile("s_wait_dscnt 0" ::: "memory");
            const unsigned nb = lds_off + (unsigned)(buf ^ 1) * 5120u;
            tdm_load_2d(nb,        Bt + (unsigned)n0 * K + k0 + 32, K, Nt, 32, 64, K);
            tdm_load_2d(nb + 4096, A  + (unsigned)m0 * K + k0 + 32, K, Mt, 32, 16, K);
            __builtin_amdgcn_s_wait_tensorcnt((short)2);  // current buf complete
        } else {
            __builtin_amdgcn_s_wait_tensorcnt((short)0);
        }
        asm volatile("" ::: "memory");    // keep LDS reads below the wait

        const unsigned short* lb = lds_base + buf * 2560;   // B tile (ushorts)
        const unsigned short* la = lb + 2048;               // A tile
        FragBF a;
        a.q[0] = *(const v4u*)(la + ln * 32 + hi * 8);
        a.q[1] = *(const v4u*)(la + ln * 32 + hi * 8 + 16);
#pragma unroll
        for (int t = 0; t < 4; ++t) {
            FragBF b;
            const unsigned short* bp = lb + (t * 16 + ln) * 32 + hi * 8;
            b.q[0] = *(const v4u*)(bp);
            b.q[1] = *(const v4u*)(bp + 16);
            acc[t] = __builtin_amdgcn_wmma_f32_16x16x32_bf16(
                false, a.v, false, b.v, (short)0, acc[t], false, false);
        }
        buf ^= 1;
    }
}

// -------------------- conversion / transpose kernels -----------------------

__global__ void cvt_bf16_kernel(const float* __restrict__ src,
                                unsigned short* __restrict__ dst, int n) {
    int i = blockIdx.x * blockDim.x + threadIdx.x;
    if (i < n) dst[i] = f2bf(src[i]);
}

// src [rows][cols] fp32 -> dst [cols][rows] bf16
__global__ void transpose_bf16_kernel(const float* __restrict__ src,
                                      unsigned short* __restrict__ dst,
                                      int rows, int cols) {
    int i = blockIdx.x * blockDim.x + threadIdx.x;
    if (i < rows * cols) {
        int n = i / rows;       // dst row  = src col
        int k = i % rows;       // dst col  = src row
        dst[n * rows + k] = f2bf(src[k * cols + n]);
    }
}

// -------------------------- QKV projection GEMM ----------------------------
__global__ __launch_bounds__(128)
void qkv_gemm_kernel(const unsigned short* __restrict__ A,
                     const unsigned short* __restrict__ Bt,
                     const float* __restrict__ bias,
                     unsigned short* __restrict__ Qb,
                     unsigned short* __restrict__ Kb,
                     unsigned short* __restrict__ Vtb) {
    __shared__ __align__(16) unsigned short Tiles[4][2][2560];  // 40KB
    const int K    = 1024;
    const int lane = threadIdx.x & 31;
    const int wave = threadIdx.x >> 5;
    const int hi   = lane >> 4;
    const int ln   = lane & 15;
    const int m0   = blockIdx.x * 32  + (wave >> 1) * 16;
    const int n0   = blockIdx.y * 128 + (wave & 1) * 64;

    v8f acc[4] = {};
    gemm_mainloop_tdm(A, Bt, &Tiles[wave][0][0], (unsigned)wave * 10240u,
                      m0, n0, K, 4096, 3072, ln, hi, acc);

#pragma unroll
    for (int t = 0; t < 4; ++t) {
#pragma unroll
        for (int r = 0; r < 8; ++r) {
            const int gm = m0 + r + 8 * hi;          // token 0..4095
            const int gn = n0 + t * 16 + ln;         // channel 0..3071
            const float v = acc[t][r] + bias[gn];
            const int bb = gm >> 10, tt = gm & 1023;
            const int sec = gn >> 10, cn = gn & 1023;
            const int h = cn >> 6,   d  = cn & 63;
            const unsigned short bv = f2bf(v);
            if (sec == 0)      Qb [((bb * 16 + h) * 1024 + tt) * 64 + d] = bv;
            else if (sec == 1) Kb [((bb * 16 + h) * 1024 + tt) * 64 + d] = bv;
            else               Vtb[((bb * 16 + h) * 64 + d) * 1024 + tt] = bv;
        }
    }
}

// ----------------------- flash causal attention ----------------------------
// one wave = one (b,h, 16-row q tile); online softmax; all matmuls via WMMA.
__global__ __launch_bounds__(128)
void attn_kernel(const unsigned short* __restrict__ Qb,
                 const unsigned short* __restrict__ Kb,
                 const unsigned short* __restrict__ Vtb,
                 unsigned short* __restrict__ Yb) {
    __shared__ __align__(16) float Plds[4][16][32];   // per-wave 2KB slice
    const int lane = threadIdx.x & 31;
    const int wave = threadIdx.x >> 5;
    const int hi   = lane >> 4;
    const int ln   = lane & 15;
    const int w    = blockIdx.x * 4 + wave;   // 0..4095
    const int bh   = w >> 6;                  // 0..63  (b*16+h)
    const int q0   = (w & 63) * 16;

    const unsigned short* Qp = Qb  + bh * 1024 * 64;
    const unsigned short* Kp = Kb  + bh * 1024 * 64;
    const unsigned short* Vp = Vtb + bh * 64 * 1024;
    float (*P)[32] = Plds[wave];
    escape(&Plds[wave][0][0]);   // cross-lane LDS round-trip: keep ds ops real

    FragBF aQ[2];
#pragma unroll
    for (int kk = 0; kk < 2; ++kk)
        aQ[kk] = load_frag_row(Qp, q0 + ln, 64, kk * 32, hi);

    v8f acc[4] = {};
    float m[8], l[8];
#pragma unroll
    for (int r = 0; r < 8; ++r) { m[r] = -1e30f; l[r] = 0.f; }

    const int nch = (q0 + 16 + 31) >> 5;       // causal extent in 32-key chunks
    for (int j = 0; j < nch; ++j) {
        const int key0 = j * 32;

        // ---- hoist ALL loads for this chunk (8 x 2 b128) -------------------
        FragBF bK[4];                           // [kk*2 + half16]
#pragma unroll
        for (int kk = 0; kk < 2; ++kk) {
            bK[2 * kk + 0] = load_frag_row(Kp, key0 + ln,      64, kk * 32, hi);
            bK[2 * kk + 1] = load_frag_row(Kp, key0 + 16 + ln, 64, kk * 32, hi);
        }
        FragBF bV[4];                           // V^T: lane = d column
#pragma unroll
        for (int t = 0; t < 4; ++t)
            bV[t] = load_frag_row(Vp, t * 16 + ln, 1024, key0, hi);

        // ---- scores: S[16 x 32] = Q @ K^T ----------------------------------
        v8f S0 = {}, S1 = {};
#pragma unroll
        for (int kk = 0; kk < 2; ++kk) {
            S0 = __builtin_amdgcn_wmma_f32_16x16x32_bf16(
                false, aQ[kk].v, false, bK[2 * kk + 0].v, (short)0, S0, false, false);
            S1 = __builtin_amdgcn_wmma_f32_16x16x32_bf16(
                false, aQ[kk].v, false, bK[2 * kk + 1].v, (short)0, S1, false, false);
        }

        // ---- causal mask + online softmax (hides V load latency) -----------
        const int kp0 = key0 + ln, kp1 = kp0 + 16;
#pragma unroll
        for (int r = 0; r < 8; ++r) {
            const int row = q0 + r + 8 * hi;
            float s0 = (kp0 <= row) ? S0[r] * 0.125f : -1e30f;   // 1/sqrt(64)
            float s1 = (kp1 <= row) ? S1[r] * 0.125f : -1e30f;
            float tmax = fmaxf(s0, s1);
#pragma unroll
            for (int o = 1; o < 16; o <<= 1) tmax = fmaxf(tmax, __shfl_xor(tmax, o, 32));
            const float mn   = fmaxf(m[r], tmax);
            const float cfac = __expf(m[r] - mn);
            const float p0 = __expf(s0 - mn);
            const float p1 = __expf(s1 - mn);
            float rs = p0 + p1;
#pragma unroll
            for (int o = 1; o < 16; o <<= 1) rs += __shfl_xor(rs, o, 32);
            l[r] = l[r] * cfac + rs;
            m[r] = mn;
#pragma unroll
            for (int t = 0; t < 4; ++t) acc[t][r] *= cfac;
            P[r + 8 * hi][ln]      = p0;       // stage P for layout transpose
            P[r + 8 * hi][16 + ln] = p1;
        }
        asm volatile("s_wait_dscnt 0" ::: "memory");   // LDS writes visible
        FragBF aP;                                     // C-layout -> A-layout
        {
            const float* pr = &P[ln][hi * 8];
            v4f f0 = *(const v4f*)(pr);
            v4f f1 = *(const v4f*)(pr + 4);
            v4f f2 = *(const v4f*)(pr + 16);
            v4f f3 = *(const v4f*)(pr + 20);
#pragma unroll
            for (int e = 0; e < 4; ++e) {
                aP.u[e]      = f2bf(f0[e]);
                aP.u[4 + e]  = f2bf(f1[e]);
                aP.u[8 + e]  = f2bf(f2[e]);
                aP.u[12 + e] = f2bf(f3[e]);
            }
        }
        asm volatile("" ::: "memory");   // keep next chunk's LDS writes after reads

        // ---- Y += P @ V  (V fragments already resident) --------------------
#pragma unroll
        for (int t = 0; t < 4; ++t)
            acc[t] = __builtin_amdgcn_wmma_f32_16x16x32_bf16(
                false, aP.v, false, bV[t].v, (short)0, acc[t], false, false);
    }

    const int bb = bh >> 4, h = bh & 15;
#pragma unroll
    for (int r = 0; r < 8; ++r) {
        const float rinv = 1.0f / l[r];
        const int row = q0 + r + 8 * hi;
        unsigned short* yrow = Yb + (bb * 1024 + row) * 1024 + h * 64;
#pragma unroll
        for (int t = 0; t < 4; ++t)
            yrow[t * 16 + ln] = f2bf(acc[t][r] * rinv);
    }
}

// -------------------------- output projection ------------------------------
__global__ __launch_bounds__(128)
void proj_gemm_kernel(const unsigned short* __restrict__ A,    // Y [4096][1024]
                      const unsigned short* __restrict__ Bt,   // Wproj^T [1024][1024]
                      const float* __restrict__ bias,
                      float* __restrict__ out) {
    __shared__ __align__(16) unsigned short Tiles[4][2][2560];  // 40KB
    const int K    = 1024;
    const int lane = threadIdx.x & 31;
    const int wave = threadIdx.x >> 5;
    const int hi   = lane >> 4;
    const int ln   = lane & 15;
    const int m0   = blockIdx.x * 32  + (wave >> 1) * 16;
    const int n0   = blockIdx.y * 128 + (wave & 1) * 64;

    v8f acc[4] = {};
    gemm_mainloop_tdm(A, Bt, &Tiles[wave][0][0], (unsigned)wave * 10240u,
                      m0, n0, K, 4096, 1024, ln, hi, acc);

#pragma unroll
    for (int t = 0; t < 4; ++t) {
#pragma unroll
        for (int r = 0; r < 8; ++r) {
            const int gm = m0 + r + 8 * hi;
            const int gn = n0 + t * 16 + ln;
            out[gm * 1024 + gn] = acc[t][r] + bias[gn];
        }
    }
}

// ------------------------------- launcher ----------------------------------
extern "C" void kernel_launch(void* const* d_in, const int* in_sizes, int n_in,
                              void* d_out, int out_size, void* d_ws, size_t ws_size,
                              hipStream_t stream) {
    const float* x     = (const float*)d_in[0];   // [4,1024,1024]
    const float* Wqkv  = (const float*)d_in[1];   // [1024,3072]
    const float* bqkv  = (const float*)d_in[2];   // [3072]
    const float* Wproj = (const float*)d_in[3];   // [1024,1024]
    const float* bproj = (const float*)d_in[4];   // [1024]
    float* out = (float*)d_out;

    char* ws = (char*)d_ws;
    unsigned short* xbf    = (unsigned short*)(ws);              //  8.0 MiB
    unsigned short* WqkvT  = (unsigned short*)(ws +  8388608);   //  6.0 MiB
    unsigned short* WprojT = (unsigned short*)(ws + 14680064);   //  2.0 MiB
    unsigned short* Qb     = (unsigned short*)(ws + 16777216);   //  8.0 MiB
    unsigned short* Kb     = (unsigned short*)(ws + 25165824);   //  8.0 MiB
    unsigned short* Vtb    = (unsigned short*)(ws + 33554432);   //  8.0 MiB
    unsigned short* Yb     = (unsigned short*)(ws + 41943040);   //  8.0 MiB

    cvt_bf16_kernel<<<(4194304 + 255) / 256, 256, 0, stream>>>(x, xbf, 4194304);
    transpose_bf16_kernel<<<(3145728 + 255) / 256, 256, 0, stream>>>(Wqkv, WqkvT, 1024, 3072);
    transpose_bf16_kernel<<<(1048576 + 255) / 256, 256, 0, stream>>>(Wproj, WprojT, 1024, 1024);

    dim3 gq(128, 24);                                 // M=4096/32, N=3072/128
    qkv_gemm_kernel<<<gq, 128, 0, stream>>>(xbf, WqkvT, bqkv, Qb, Kb, Vtb);

    attn_kernel<<<1024, 128, 0, stream>>>(Qb, Kb, Vtb, Yb);   // 4096 waves

    dim3 gp(128, 8);                                  // M=4096/32, N=1024/128
    proj_gemm_kernel<<<gp, 128, 0, stream>>>(Yb, WprojT, bproj, out);
}
